// DCSA_23897198035617
// MI455X (gfx1250) — compile-verified, matrix operand order checked by reading, not run on hardware
//
#include <hip/hip_runtime.h>
#include <hip/hip_bf16.h>

// ---------------------------------------------------------------------------
// DCSA block for MI455X (gfx1250, wave32, WMMA).
// All GEMM-shaped math runs through v_wmma_f32_16x16x32_f16 (f16 in, f32 acc).
// LDS tiles are stored so every WMMA fragment loads as 2x ds_load_b128.
// GEMM/conv: 128x64 tile, each wave = 32 rows x 64 cols = 8 WMMA / K-step.
// ---------------------------------------------------------------------------

typedef __attribute__((ext_vector_type(16))) _Float16 v16h;
typedef __attribute__((ext_vector_type(8)))  _Float16 v8h;
typedef __attribute__((ext_vector_type(4)))  _Float16 v4h;
typedef __attribute__((ext_vector_type(8)))  float    v8f;
typedef __attribute__((ext_vector_type(4)))  float    v4f;

#define WAVE (threadIdx.x >> 5)
#define LANE (threadIdx.x & 31)

// Unified fragment loader (A and transposed-B share the same lane layout):
// lane l (0-15): row = l, half-chunks at K {kb..kb+7} and {kb+16..kb+23},
// kb = 0 for lanes 0-15, 8 for lanes 16-31.  Two 16B contiguous LDS loads.
__device__ __forceinline__ v16h load_frag(const _Float16* p, int stride) {
  const int lane = LANE;
  const _Float16* r = p + (lane & 15) * stride + ((lane & 16) ? 8 : 0);
  v8h lo = *(const v8h*)r;
  v8h hi = *(const v8h*)(r + 16);
  return __builtin_shufflevector(lo, hi, 0, 1, 2, 3, 4, 5, 6, 7,
                                 8, 9, 10, 11, 12, 13, 14, 15);
}
__device__ __forceinline__ v8f wmma16(v16h a, v16h b, v8f c) {
  return __builtin_amdgcn_wmma_f32_16x16x32_f16(false, a, false, b, (short)0, c,
                                                false, false);
}
__device__ __forceinline__ v4h cvt4(float a, float b, float c, float d) {
  v4h h = {(_Float16)a, (_Float16)b, (_Float16)c, (_Float16)d};
  return h;
}

// LDS row strides (halves): padded, 16B-aligned, bank-conflict friendly.
#define S32 40   // tiles with 32-element rows
#define S64 72   // tiles with 64-element rows

// --- Generic WMMA GEMM:  out[b, coff+m, hw] = (R?) + sum_k W[m,k]*X[b,k,hw] --
// X layout [B, K, HW=1024]; W row-major [M, K]; 128x64 tile, 128 threads.
__global__ __launch_bounds__(128) void wmma_gemm_kernel(
    const float* __restrict__ W, const float* __restrict__ X,
    const float* __restrict__ R, float* __restrict__ out,
    int K, int out_bstride, int out_coff, int r_bstride) {
  __shared__ alignas(16) _Float16 lA[128 * S32];  // [m][k]
  __shared__ alignas(16) _Float16 lBT[64 * S32];  // [n][k] (transposed)
  const int tid = threadIdx.x;
  const int n0 = blockIdx.x * 64, m0 = blockIdx.y * 128;
  v8f acc0[4] = {v8f{}, v8f{}, v8f{}, v8f{}};
  v8f acc1[4] = {v8f{}, v8f{}, v8f{}, v8f{}};
  for (int k0 = 0; k0 < K; k0 += 32) {
    __syncthreads();
#pragma unroll
    for (int j = 0; j < 8; ++j) {                 // A: 128x32, float4 -> b64
      int idx = tid + 128 * j;
      int m = idx >> 3, g = idx & 7;
      const float* wp = &W[(size_t)(m0 + m) * K + k0 + g * 4];
      __builtin_prefetch(wp + 32, 0, 1);          // speculative, next K-tile
      v4f w4 = *(const v4f*)wp;
      *(v4h*)&lA[m * S32 + g * 4] = cvt4(w4[0], w4[1], w4[2], w4[3]);
    }
#pragma unroll
    for (int j = 0; j < 4; ++j) {                 // B^T: [n][k]
      int idx = tid + 128 * j;
      int n = idx >> 3, g = idx & 7;
      int gn = n0 + n, bb = gn >> 10, hw = gn & 1023;
      const float* xp = &X[((size_t)bb * K + (k0 + g * 4)) * 1024 + hw];
      __builtin_prefetch(xp + 32 * 1024, 0, 1);   // next K-tile of activations
      *(v4h*)&lBT[n * S32 + g * 4] = cvt4(xp[0], xp[1024], xp[2048], xp[3072]);
    }
    __syncthreads();
    v16h a0 = load_frag(lA + (WAVE * 32) * S32, S32);
    v16h a1 = load_frag(lA + (WAVE * 32 + 16) * S32, S32);
    v16h bf[4];
#pragma unroll
    for (int t = 0; t < 4; ++t) bf[t] = load_frag(lBT + (t * 16) * S32, S32);
#pragma unroll
    for (int t = 0; t < 4; ++t) {
      acc0[t] = wmma16(a0, bf[t], acc0[t]);
      acc1[t] = wmma16(a1, bf[t], acc1[t]);
    }
  }
  const int half = LANE >> 4, col = LANE & 15;
#pragma unroll
  for (int aa = 0; aa < 2; ++aa)
#pragma unroll
    for (int t = 0; t < 4; ++t)
#pragma unroll
      for (int v = 0; v < 8; ++v) {
        int m = m0 + WAVE * 32 + aa * 16 + v + 8 * half;
        int gn = n0 + t * 16 + col, b = gn >> 10, hw = gn & 1023;
        float val = aa ? acc1[t][v] : acc0[t][v];
        if (R) val += R[(size_t)b * r_bstride + (size_t)m * 1024 + hw];
        out[(size_t)b * out_bstride + (size_t)(out_coff + m) * 1024 + hw] = val;
      }
}

// --- Implicit-GEMM conv (KSxKS, pad): im2col gather fused into B^T staging --
template <int KS, int PAD>
__global__ __launch_bounds__(128) void wmma_conv_kernel(
    const float* __restrict__ W,   // [256, 256, KS, KS] == [M, Ktot] row-major
    const float* __restrict__ X,   // [8, 256, 32, 32]
    float* __restrict__ out) {     // [8, 256, 32, 32]
  constexpr int Ktot = 256 * KS * KS;
  __shared__ alignas(16) _Float16 lA[128 * S32];
  __shared__ alignas(16) _Float16 lBT[64 * S32];
  const int tid = threadIdx.x;
  const int n0 = blockIdx.x * 64, m0 = blockIdx.y * 128;
  v8f acc0[4] = {v8f{}, v8f{}, v8f{}, v8f{}};
  v8f acc1[4] = {v8f{}, v8f{}, v8f{}, v8f{}};
  for (int k0 = 0; k0 < Ktot; k0 += 32) {
    __syncthreads();
#pragma unroll
    for (int j = 0; j < 8; ++j) {
      int idx = tid + 128 * j;
      int m = idx >> 3, g = idx & 7;
      const float* wp = &W[(size_t)(m0 + m) * Ktot + k0 + g * 4];
      __builtin_prefetch(wp + 32, 0, 1);
      v4f w4 = *(const v4f*)wp;
      *(v4h*)&lA[m * S32 + g * 4] = cvt4(w4[0], w4[1], w4[2], w4[3]);
    }
#pragma unroll
    for (int j = 0; j < 4; ++j) {
      int idx = tid + 128 * j;
      int n = idx >> 3, g = idx & 7;
      int gn = n0 + n, bb = gn >> 10, hw = gn & 1023;
      int yy = hw >> 5, xx = hw & 31;
      v4h h4;
#pragma unroll
      for (int q = 0; q < 4; ++q) {
        int kk = k0 + g * 4 + q;
        int ci = kk / (KS * KS), t = kk % (KS * KS);
        int y = yy + t / KS - PAD, x = xx + t % KS - PAD;
        float val = 0.f;
        if ((unsigned)y < 32u && (unsigned)x < 32u)
          val = X[((size_t)bb * 256 + ci) * 1024 + y * 32 + x];
        h4[q] = (_Float16)val;
      }
      *(v4h*)&lBT[n * S32 + g * 4] = h4;
    }
    __syncthreads();
    v16h a0 = load_frag(lA + (WAVE * 32) * S32, S32);
    v16h a1 = load_frag(lA + (WAVE * 32 + 16) * S32, S32);
    v16h bf[4];
#pragma unroll
    for (int t = 0; t < 4; ++t) bf[t] = load_frag(lBT + (t * 16) * S32, S32);
#pragma unroll
    for (int t = 0; t < 4; ++t) {
      acc0[t] = wmma16(a0, bf[t], acc0[t]);
      acc1[t] = wmma16(a1, bf[t], acc1[t]);
    }
  }
  const int half = LANE >> 4, col = LANE & 15;
#pragma unroll
  for (int aa = 0; aa < 2; ++aa)
#pragma unroll
    for (int t = 0; t < 4; ++t)
#pragma unroll
      for (int v = 0; v < 8; ++v) {
        int m = m0 + WAVE * 32 + aa * 16 + v + 8 * half;
        int gn = n0 + t * 16 + col, b = gn >> 10, hw = gn & 1023;
        out[((size_t)b * 256 + m) * 1024 + hw] = aa ? acc1[t][v] : acc0[t][v];
      }
}

// --- LayerNorm over C=256 (NCHW channel dim), one wave per pixel ------------
__global__ __launch_bounds__(256) void layernorm_kernel(
    const float* __restrict__ x, const float* __restrict__ w,
    const float* __restrict__ bia, float* __restrict__ y) {
  int pix = blockIdx.x * 8 + WAVE;          // 8192 pixels total
  int lane = LANE;
  int b = pix >> 10, hw = pix & 1023;
  const float* px = x + (size_t)b * 256 * 1024 + hw;
  float vals[8], s = 0.f, s2 = 0.f;
#pragma unroll
  for (int j = 0; j < 8; ++j) {
    float v = px[(size_t)(lane + 32 * j) * 1024];
    vals[j] = v; s += v; s2 += v * v;
  }
#pragma unroll
  for (int m = 1; m < 32; m <<= 1) {
    s += __shfl_xor(s, m, 32);
    s2 += __shfl_xor(s2, m, 32);
  }
  float mean = s * (1.f / 256.f);
  float var = s2 * (1.f / 256.f) - mean * mean;
  float inv = rsqrtf(var + 1e-5f);
  float* py = y + (size_t)b * 256 * 1024 + hw;
#pragma unroll
  for (int j = 0; j < 8; ++j) {
    int c = lane + 32 * j;
    py[(size_t)c * 1024] = (vals[j] - mean) * inv * w[c] + bia[c];
  }
}

// --- Flash MHSA: heads=4, d=64, N=1024. One WG per (b,h,64-query block). ----
__global__ __launch_bounds__(128) void mhsa_kernel(
    const float* __restrict__ qkv,   // [B, 768, 1024]
    float* __restrict__ outp) {      // [B, 256, 1024]
  __shared__ alignas(16) _Float16 lq[64 * S64];      // [q][d]
  __shared__ alignas(16) _Float16 lk[32 * S64];      // [tok][d]  (B^T for S)
  __shared__ alignas(16) _Float16 lv[64 * S32];      // [d][tok]  (B^T for O)
  __shared__ alignas(16) _Float16 lp[4 * 16 * S32];  // per-wave P [q][k]
  const int tid = threadIdx.x, wave = WAVE, lane = LANE;
  const int q0 = blockIdx.x * 64, h = blockIdx.y, b = blockIdx.z;
  const size_t base = (size_t)b * 768 * 1024;
  const float* Q = qkv + base + (size_t)(h * 64) * 1024;
  const float* Kp = qkv + base + (size_t)(256 + h * 64) * 1024;
  const float* Vp = qkv + base + (size_t)(512 + h * 64) * 1024;
#pragma unroll
  for (int j = 0; j < 8; ++j) {         // Q tile [q][d] (source transposed)
    int idx = tid + 128 * j;
    int q = idx >> 4, g = idx & 15;
    const float* qp = &Q[(size_t)(g * 4) * 1024 + q0 + q];
    *(v4h*)&lq[q * S64 + g * 4] = cvt4(qp[0], qp[1024], qp[2048], qp[3072]);
  }
  v8f o[4] = {v8f{}, v8f{}, v8f{}, v8f{}};
  float rmax[8], rsum[8];
#pragma unroll
  for (int v = 0; v < 8; ++v) { rmax[v] = -1e30f; rsum[v] = 0.f; }
  const int half = lane >> 4, col = lane & 15;
  _Float16* myP = lp + wave * 16 * S32;

  for (int kb = 0; kb < 1024; kb += 32) {
    __syncthreads();
#pragma unroll
    for (int j = 0; j < 4; ++j) {       // K tile [tok][d]
      int idx = tid + 128 * j;
      int t = idx >> 4, g = idx & 15;
      const float* kp = &Kp[(size_t)(g * 4) * 1024 + kb + t];
      __builtin_prefetch(kp + 32, 0, 1);
      *(v4h*)&lk[t * S64 + g * 4] = cvt4(kp[0], kp[1024], kp[2048], kp[3072]);
    }
#pragma unroll
    for (int j = 0; j < 4; ++j) {       // V tile [d][tok] (contiguous src)
      int idx = tid + 128 * j;
      int d = idx >> 3, g = idx & 7;
      const float* vp = &Vp[(size_t)d * 1024 + kb + g * 4];
      __builtin_prefetch(vp + 32, 0, 1);
      v4f v4 = *(const v4f*)vp;
      *(v4h*)&lv[d * S32 + g * 4] = cvt4(v4[0], v4[1], v4[2], v4[3]);
    }
    __syncthreads();
    // S = (Q K^T) * d^-0.5 : 16q x 32k per wave
    v8f s0 = v8f{}, s1 = v8f{};
    {
      v16h a0 = load_frag(lq + (wave * 16) * S64, S64);
      v16h a1 = load_frag(lq + (wave * 16) * S64 + 32, S64);
      v16h b00 = load_frag(lk, S64);                  // tok 0-15,  d 0-31
      v16h b10 = load_frag(lk + 16 * S64, S64);       // tok 16-31, d 0-31
      v16h b01 = load_frag(lk + 32, S64);             // tok 0-15,  d 32-63
      v16h b11 = load_frag(lk + 16 * S64 + 32, S64);  // tok 16-31, d 32-63
      s0 = wmma16(a0, b00, s0);
      s1 = wmma16(a0, b10, s1);
      s0 = wmma16(a1, b01, s0);
      s1 = wmma16(a1, b11, s1);
    }
    // online softmax (rows live across 16-lane halves)
    float bm[8];
#pragma unroll
    for (int v = 0; v < 8; ++v) {
      s0[v] *= 0.125f; s1[v] *= 0.125f;
      bm[v] = fmaxf(s0[v], s1[v]);
    }
#pragma unroll
    for (int m = 1; m < 16; m <<= 1)
#pragma unroll
      for (int v = 0; v < 8; ++v) bm[v] = fmaxf(bm[v], __shfl_xor(bm[v], m, 32));
    float corr[8], ps[8];
#pragma unroll
    for (int v = 0; v < 8; ++v) {
      float nm = fmaxf(rmax[v], bm[v]);
      corr[v] = __expf(rmax[v] - nm);
      rmax[v] = nm;
      float p0 = __expf(s0[v] - nm), p1 = __expf(s1[v] - nm);
      s0[v] = p0; s1[v] = p1;
      ps[v] = p0 + p1;
    }
#pragma unroll
    for (int m = 1; m < 16; m <<= 1)
#pragma unroll
      for (int v = 0; v < 8; ++v) ps[v] += __shfl_xor(ps[v], m, 32);
#pragma unroll
    for (int v = 0; v < 8; ++v) rsum[v] = rsum[v] * corr[v] + ps[v];
#pragma unroll
    for (int t = 0; t < 4; ++t)
#pragma unroll
      for (int v = 0; v < 8; ++v) o[t][v] *= corr[v];
    // C-layout -> A-layout via per-wave LDS bounce
#pragma unroll
    for (int v = 0; v < 8; ++v) {
      int r = v + 8 * half;
      myP[r * S32 + col] = (_Float16)s0[v];
      myP[r * S32 + 16 + col] = (_Float16)s1[v];
    }
    asm volatile("s_wait_dscnt 0" ::: "memory");   // wave-local LDS RAW
    v16h ap = load_frag(myP, S32);
    v16h bv[4];
#pragma unroll
    for (int t = 0; t < 4; ++t) bv[t] = load_frag(lv + (t * 16) * S32, S32);
#pragma unroll
    for (int t = 0; t < 4; ++t) o[t] = wmma16(ap, bv[t], o[t]);
  }
  // normalize + store (out is [B, C, N], channel-major like inputs)
#pragma unroll
  for (int t = 0; t < 4; ++t)
#pragma unroll
    for (int v = 0; v < 8; ++v) {
      float val = o[t][v] / rsum[v];
      int ch = h * 64 + t * 16 + col;
      int n = q0 + wave * 16 + v + 8 * half;
      outp[((size_t)b * 256 + ch) * 1024 + n] = val;
    }
}

// --- Depthwise 3x3 conv + exact GELU (LeFF middle), pure VALU ---------------
__global__ __launch_bounds__(256) void dwconv_gelu_kernel(
    const float* __restrict__ x,   // [8, 1024, 32, 32]
    const float* __restrict__ w,   // [1024, 1, 3, 3]
    float* __restrict__ y) {
  size_t i = (size_t)blockIdx.x * 256 + threadIdx.x;
  int hw = (int)(i & 1023);
  int c = (int)((i >> 10) & 1023);
  int b = (int)(i >> 20);
  int yy = hw >> 5, xx = hw & 31;
  const float* px = x + ((size_t)b * 1024 + c) * 1024;
  const float* pw = w + c * 9;
  float s = 0.f;
#pragma unroll
  for (int dy = -1; dy <= 1; ++dy)
#pragma unroll
    for (int dx = -1; dx <= 1; ++dx) {
      int y2 = yy + dy, x2 = xx + dx;
      if ((unsigned)y2 < 32u && (unsigned)x2 < 32u)
        s += pw[(dy + 1) * 3 + (dx + 1)] * px[y2 * 32 + x2];
    }
  y[i] = 0.5f * s * (1.f + erff(s * 0.70710678118654752f));
}

// ---------------------------------------------------------------------------
extern "C" void kernel_launch(void* const* d_in, const int* in_sizes, int n_in,
                              void* d_out, int out_size, void* d_ws,
                              size_t ws_size, hipStream_t stream) {
  (void)in_sizes; (void)n_in; (void)out_size; (void)ws_size;
  const float* aop    = (const float*)d_in[0];
  const float* dop    = (const float*)d_in[1];
  const float* wq     = (const float*)d_in[2];   // [256,256,3,3]
  const float* wkv    = (const float*)d_in[3];   // [256,256,5,5]
  const float* lnq1w  = (const float*)d_in[4];
  const float* lnq1b  = (const float*)d_in[5];
  const float* lnkv1w = (const float*)d_in[6];
  const float* lnkv1b = (const float*)d_in[7];
  const float* lnq2w  = (const float*)d_in[8];
  const float* lnq2b  = (const float*)d_in[9];
  const float* lnkv2w = (const float*)d_in[10];
  const float* lnkv2b = (const float*)d_in[11];
  const float* lnffnw = (const float*)d_in[12];
  const float* lnffnb = (const float*)d_in[13];
  const float* saq_qkv  = (const float*)d_in[14];  // [768,256]
  const float* saq_proj = (const float*)d_in[15];  // [256,256]
  const float* sakv_qkv = (const float*)d_in[16];
  const float* sakv_proj= (const float*)d_in[17];
  const float* ca_q   = (const float*)d_in[18];
  const float* ca_k   = (const float*)d_in[19];
  const float* ca_v   = (const float*)d_in[20];
  const float* ca_proj= (const float*)d_in[21];
  const float* leff_in  = (const float*)d_in[22];  // [1024,256]
  const float* leff_dw  = (const float*)d_in[23];  // [1024,1,3,3]
  const float* leff_out = (const float*)d_in[24];  // [256,1024]
  float* out = (float*)d_out;

  const size_t T = (size_t)8 * 256 * 1024;  // 2,097,152 floats per activation
  float* ws   = (float*)d_ws;
  float* qb   = ws;            // q_branch               [8,256,1024]
  float* kvb  = ws + 1 * T;    // kv_branch
  float* lnt  = ws + 2 * T;    // layernorm scratch
  float* qkvb = ws + 3 * T;    // qkv / q|k|v            [8,768,1024] (3T)
  float* attn = ws + 6 * T;    // attention output
  float* qn   = ws + 7 * T;    // LN(q_branch)
  float* kvn  = ws + 8 * T;    // LN(kv_branch)
  float* xb   = ws + 9 * T;    // x
  float* hid  = ws + 10 * T;   // LeFF hidden            [8,1024,1024] (4T)
  float* hid2 = ws + 14 * T;   // after dwconv+gelu      (4T)

  const dim3 thr128(128), thr256(256);
  const dim3 gM256(128, 2);    // M=256 (128-row tiles)
  const dim3 gM768(128, 6);    // M=768
  const dim3 gM1024(128, 8);   // M=1024
  const dim3 gLN(1024);
  const dim3 gAtt(16, 4, 8);

  // ---- q branch -----------------------------------------------------------
  wmma_conv_kernel<3, 1><<<gM256, thr128, 0, stream>>>(wq, aop, qb);
  layernorm_kernel<<<gLN, thr256, 0, stream>>>(qb, lnq1w, lnq1b, lnt);
  wmma_gemm_kernel<<<gM768, thr128, 0, stream>>>(saq_qkv, lnt, nullptr, qkvb,
                                                 256, 768 * 1024, 0, 0);
  mhsa_kernel<<<gAtt, thr128, 0, stream>>>(qkvb, attn);
  wmma_gemm_kernel<<<gM256, thr128, 0, stream>>>(saq_proj, attn, qb, qb,
                                                 256, 256 * 1024, 0, 256 * 1024);
  layernorm_kernel<<<gLN, thr256, 0, stream>>>(qb, lnq2w, lnq2b, qn);

  // ---- kv branch ----------------------------------------------------------
  wmma_conv_kernel<5, 2><<<gM256, thr128, 0, stream>>>(wkv, dop, kvb);
  layernorm_kernel<<<gLN, thr256, 0, stream>>>(kvb, lnkv1w, lnkv1b, lnt);
  wmma_gemm_kernel<<<gM768, thr128, 0, stream>>>(sakv_qkv, lnt, nullptr, qkvb,
                                                 256, 768 * 1024, 0, 0);
  mhsa_kernel<<<gAtt, thr128, 0, stream>>>(qkvb, attn);
  wmma_gemm_kernel<<<gM256, thr128, 0, stream>>>(sakv_proj, attn, kvb, kvb,
                                                 256, 256 * 1024, 0, 256 * 1024);
  layernorm_kernel<<<gLN, thr256, 0, stream>>>(kvb, lnkv2w, lnkv2b, kvn);

  // ---- cross attention ----------------------------------------------------
  wmma_gemm_kernel<<<gM256, thr128, 0, stream>>>(ca_q, qn, nullptr, qkvb,
                                                 256, 768 * 1024, 0, 0);
  wmma_gemm_kernel<<<gM256, thr128, 0, stream>>>(ca_k, kvn, nullptr, qkvb,
                                                 256, 768 * 1024, 256, 0);
  wmma_gemm_kernel<<<gM256, thr128, 0, stream>>>(ca_v, kvn, nullptr, qkvb,
                                                 256, 768 * 1024, 512, 0);
  mhsa_kernel<<<gAtt, thr128, 0, stream>>>(qkvb, attn);
  wmma_gemm_kernel<<<gM256, thr128, 0, stream>>>(ca_proj, attn, qb, xb,
                                                 256, 256 * 1024, 0, 256 * 1024);

  // ---- LeFF ---------------------------------------------------------------
  layernorm_kernel<<<gLN, thr256, 0, stream>>>(xb, lnffnw, lnffnb, lnt);
  wmma_gemm_kernel<<<gM1024, thr128, 0, stream>>>(leff_in, lnt, nullptr, hid,
                                                  256, 1024 * 1024, 0, 0);
  dwconv_gelu_kernel<<<dim3(32768), thr256, 0, stream>>>(hid, leff_dw, hid2);
  wmma_gemm_kernel<<<gM256, thr128, 0, stream>>>(leff_out, hid2, xb, out,
                                                 1024, 256 * 1024, 0, 256 * 1024);
}
